// GraphSAGE_17755394802084
// MI455X (gfx1250) — matrix operand order repaired and checked by reading, not compiled, not run on hardware
//
#include <hip/hip_runtime.h>
#include <hip/hip_bf16.h>

#define N_NODES 100000
#define E_EDGES 800000
#define F_IN    182
#define HID1    128
#define HID2    64
#define NCLS    2

typedef __attribute__((ext_vector_type(16))) __bf16 v16bf;
typedef __attribute__((ext_vector_type(8)))  __bf16 v8bf;
typedef __attribute__((ext_vector_type(8)))  float  v8f;

// ---------------------------------------------------------------------------
// Degree count: E fp32 atomics (global_atomic_add_f32, no-return -> STOREcnt)
// ---------------------------------------------------------------------------
__global__ void count_kernel(const int* __restrict__ dst, float* __restrict__ cnt) {
    int e = blockIdx.x * blockDim.x + threadIdx.x;
    if (e < E_EDGES) unsafeAtomicAdd(&cnt[dst[e]], 1.0f);
}

// ---------------------------------------------------------------------------
// Edge scatter-add: one wave32 per edge; lanes stride the feature row.
// Coalesced row gather + hardware fp32 atomics for the segment-sum.
// ---------------------------------------------------------------------------
template <int F>
__global__ void scatter_kernel(const float* __restrict__ feat,
                               const int* __restrict__ src_idx,
                               const int* __restrict__ dst_idx,
                               float* __restrict__ accum) {
    int wave = (blockIdx.x * blockDim.x + threadIdx.x) >> 5;
    int lane = threadIdx.x & 31;
    if (wave >= E_EDGES) return;
    int s = src_idx[wave];
    int d = dst_idx[wave];
    const float* sp = feat + (size_t)s * F;
    float*       dp = accum + (size_t)d * F;
    for (int f = lane; f < F; f += 32)
        unsafeAtomicAdd(&dp[f], sp[f]);
}

// ---------------------------------------------------------------------------
// Pack [Wl | Wr] (row-major [HOUT][FIN]) into bf16 B-matrix fragments in the
// exact wave32 WMMA layout: frag (ntile, chunk): lane holds col = ntile*16 +
// (lane&15); element e -> k = chunk*32 + (lane>=16 ? 8 : 0) + (e&7) + (e>=8 ? 16 : 0).
// K-space: [0,FIN) = Wl, [KHALF, KHALF+FIN) = Wr, rest zero-pad.
// One v16bf (32B) load per lane per WMMA in the GEMM.
// ---------------------------------------------------------------------------
__global__ void pack_w_kernel(const float* __restrict__ Wl,
                              const float* __restrict__ Wr,
                              __bf16* __restrict__ out,
                              int fin, int khalf, int nchunks, int total) {
    int idx = blockIdx.x * blockDim.x + threadIdx.x;
    if (idx >= total) return;
    int e    = idx & 15;
    int t    = idx >> 4;
    int lane = t & 31;  t >>= 5;
    int chunk = t % nchunks;
    int ntile = t / nchunks;
    int k = chunk * 32 + ((lane >> 4) << 3) + (e & 7) + ((e >> 3) << 4);
    int n = ntile * 16 + (lane & 15);
    float v = 0.0f;
    if (k < fin)                                  v = Wl[n * fin + k];
    else if (k >= khalf && (k - khalf) < fin)     v = Wr[n * fin + (k - khalf)];
    out[idx] = (__bf16)v;
}

// ---------------------------------------------------------------------------
// Fused SAGE linear:  out = relu( mean @ Wl.T + x @ Wr.T + b )
// as one K-concatenated bf16 WMMA GEMM (K = 2*KHALF, NCH chunks of 32).
// Block = 16 output rows; wave w owns output columns [w*16, w*16+16).
// A panel (16 x 2*KHALF bf16) staged in LDS, row stride padded (+8) so the
// 16 row-parallel ds_load_b128s spread across banks. Per-lane A fragment:
//   elements 0..7  = K[k0 .. k0+7],  elements 8..15 = K[k0+16 .. k0+23],
//   k0 = chunk*32 + (lane>=16 ? 8 : 0), row = lane&15   (ISA 7.12.2 layout)
// ---------------------------------------------------------------------------
template <int FIN, int KHALF, int NCH, int HOUT, int NWAVE>
__global__ __launch_bounds__(NWAVE * 32)
void sage_gemm_kernel(const float* __restrict__ ssum,
                      const float* __restrict__ cnt,
                      const float* __restrict__ xfeat,
                      const __bf16* __restrict__ wpack,
                      const float* __restrict__ bias,
                      float* __restrict__ out) {
    constexpr int KTOT = 2 * KHALF;
    constexpr int LDP  = KTOT + 8;             // padded LDS row stride (bf16)
    __shared__ __align__(16) __bf16 apanel[16][LDP];
    __shared__ float inv_s[16];

    const int m0  = blockIdx.x * 16;           // N_NODES % 16 == 0
    const int tid = threadIdx.x;

    if (tid < 16) inv_s[tid] = 1.0f / fmaxf(cnt[m0 + tid], 1.0f);
    __syncthreads();

    // Stage A panel: [mean*inv | x], bf16, zero-padded between halves.
    for (int idx = tid; idx < 16 * KTOT; idx += NWAVE * 32) {
        int row = idx / KTOT;
        int k   = idx % KTOT;
        int gr  = m0 + row;
        float v = 0.0f;
        if (k < FIN)                               v = ssum[(size_t)gr * FIN + k] * inv_s[row];
        else if (k >= KHALF && (k - KHALF) < FIN)  v = xfeat[(size_t)gr * FIN + (k - KHALF)];
        apanel[row][k] = (__bf16)v;
    }
    __syncthreads();

    const int wave = tid >> 5;                 // = output n-tile
    const int lane = tid & 31;
    const int half = lane >> 4;
    const int lrow = lane & 15;

    v8f acc = {};
    const __bf16* wp = wpack + ((size_t)wave * NCH * 32 + lane) * 16;

    #pragma unroll
    for (int c = 0; c < NCH; ++c) {
        int k0 = c * 32 + half * 8;
        v8bf alo = *(const v8bf*)&apanel[lrow][k0];
        v8bf ahi = *(const v8bf*)&apanel[lrow][k0 + 16];
        v16bf a;
        #pragma unroll
        for (int i = 0; i < 8; ++i) { a[i] = alo[i]; a[8 + i] = ahi[i]; }
        v16bf b = *(const v16bf*)(wp + (size_t)c * 32 * 16);
        acc = __builtin_amdgcn_wmma_f32_16x16x32_bf16(
                  false, a, false, b, (short)0, acc, false, false);
    }

    // Epilogue: bias + ReLU.  D layout: vgpr r -> row m0 + r + half*8, col lane&15.
    const int col = wave * 16 + lrow;
    const float bb = bias[col];
    #pragma unroll
    for (int r = 0; r < 8; ++r) {
        int gr = m0 + r + half * 8;
        float v = acc[r] + bb;
        out[(size_t)gr * HOUT + col] = v > 0.0f ? v : 0.0f;
    }
}

// ---------------------------------------------------------------------------
// Head: logits = h2 @ Wlin.T + b, then log_softmax over C=2.
// ---------------------------------------------------------------------------
__global__ void head_kernel(const float* __restrict__ h2,
                            const float* __restrict__ wlin,
                            const float* __restrict__ blin,
                            float* __restrict__ out) {
    __shared__ __align__(16) float w_s[NCLS * HID2];
    __shared__ float b_s[NCLS];
    int tid = threadIdx.x;
    if (tid < NCLS * HID2) w_s[tid] = wlin[tid];
    if (tid < NCLS)        b_s[tid] = blin[tid];
    __syncthreads();

    int n = blockIdx.x * blockDim.x + tid;
    if (n >= N_NODES) return;

    const float4* hp = (const float4*)(h2 + (size_t)n * HID2);
    const float4* w0 = (const float4*)(w_s);
    const float4* w1 = (const float4*)(w_s + HID2);
    float l0 = b_s[0], l1 = b_s[1];
    #pragma unroll
    for (int i = 0; i < HID2 / 4; ++i) {
        float4 h = hp[i];
        float4 a = w0[i];
        float4 b = w1[i];
        l0 += h.x * a.x + h.y * a.y + h.z * a.z + h.w * a.w;
        l1 += h.x * b.x + h.y * b.y + h.z * b.z + h.w * b.w;
    }
    float m   = fmaxf(l0, l1);
    float lse = m + __logf(__expf(l0 - m) + __expf(l1 - m));
    out[(size_t)n * NCLS + 0] = l0 - lse;
    out[(size_t)n * NCLS + 1] = l1 - lse;
}

// ---------------------------------------------------------------------------
extern "C" void kernel_launch(void* const* d_in, const int* in_sizes, int n_in,
                              void* d_out, int out_size, void* d_ws, size_t ws_size,
                              hipStream_t stream) {
    const float* x    = (const float*)d_in[0];
    const int*   ei   = (const int*)  d_in[1];
    const float* W1l  = (const float*)d_in[2];
    const float* W1r  = (const float*)d_in[3];
    const float* b1   = (const float*)d_in[4];
    const float* W2l  = (const float*)d_in[5];
    const float* W2r  = (const float*)d_in[6];
    const float* b2   = (const float*)d_in[7];
    const float* Wlin = (const float*)d_in[8];
    const float* blin = (const float*)d_in[9];
    const int* src = ei;
    const int* dst = ei + E_EDGES;

    // Workspace layout (~150 MB)
    float* ws   = (float*)d_ws;
    float* cnt  = ws;                                   // N
    float* sum1 = cnt  + N_NODES;                       // N * F_IN
    float* h1   = sum1 + (size_t)N_NODES * F_IN;        // N * HID1
    float* h2   = h1   + (size_t)N_NODES * HID1;        // N * HID2
    __bf16* pw1 = (__bf16*)(h2 + (size_t)N_NODES * HID2);   // 8*12*32*16 bf16
    __bf16* pw2 = pw1 + 8 * 12 * 32 * 16;                   // 4*8*32*16  bf16
    float* sum2 = sum1;   // alias: sum1 is dead once layer-1 GEMM completed

    constexpr int PW1 = 8 * 12 * 32 * 16;   // 49152 frags  (HID1/16 tiles x 12 chunks)
    constexpr int PW2 = 4 * 8 * 32 * 16;    // 16384 frags  (HID2/16 tiles x 8 chunks)

    // ---- Layer 1 ----
    hipMemsetAsync(cnt,  0, (size_t)N_NODES * sizeof(float), stream);
    hipMemsetAsync(sum1, 0, (size_t)N_NODES * F_IN * sizeof(float), stream);
    count_kernel<<<(E_EDGES + 255) / 256, 256, 0, stream>>>(dst, cnt);
    scatter_kernel<F_IN><<<E_EDGES / 8, 256, 0, stream>>>(x, src, dst, sum1);
    pack_w_kernel<<<(PW1 + 255) / 256, 256, 0, stream>>>(W1l, W1r, pw1, F_IN, 192, 12, PW1);
    sage_gemm_kernel<F_IN, 192, 12, HID1, 8>
        <<<N_NODES / 16, 256, 0, stream>>>(sum1, cnt, x, pw1, b1, h1);

    // ---- Layer 2 ----
    hipMemsetAsync(sum2, 0, (size_t)N_NODES * HID1 * sizeof(float), stream);
    scatter_kernel<HID1><<<E_EDGES / 8, 256, 0, stream>>>(h1, src, dst, sum2);
    pack_w_kernel<<<(PW2 + 255) / 256, 256, 0, stream>>>(W2l, W2r, pw2, HID1, 128, 8, PW2);
    sage_gemm_kernel<HID1, 128, 8, HID2, 4>
        <<<N_NODES / 16, 128, 0, stream>>>(sum2, cnt, h1, pw2, b2, h2);

    // ---- Head ----
    head_kernel<<<(N_NODES + 255) / 256, 256, 0, stream>>>(h2, Wlin, blin, (float*)d_out);
}
